// Bert_13889924235282
// MI455X (gfx1250) — matrix operand order repaired and checked by reading, not compiled, not run on hardware
//
#include <hip/hip_runtime.h>

// ---------------- problem constants (BERT-base, from reference) ----------------
#define BB 16
#define SS 512
#define DD 768
#define HH 12
#define KK 64
#define VV 64
#define FF 3072
#define LL 12
#define MM (BB * SS)   /* 8192 tokens */
#define HK (HH * KK)   /* 768 */
#define EPSF 1e-12f

typedef __attribute__((ext_vector_type(16))) __bf16   v16bf;
typedef __attribute__((ext_vector_type(8)))  float    v8f;
typedef __attribute__((ext_vector_type(4)))  unsigned int v4u;

static __device__ __forceinline__ unsigned short f2bf(float f) {
  union { float f; unsigned int u; } x; x.f = f;
  unsigned int r = (x.u + 0x7fffu + ((x.u >> 16) & 1u)) >> 16;  // RNE
  return (unsigned short)r;
}

static __device__ __forceinline__ float gelu_exact(float x) {
  return 0.5f * x * (1.0f + erff(x * 0.70710678118654752f));
}

// ------- weight conversion f32 -> bf16 with transpose: src[K,N] -> dst[N,K] -------
__global__ __launch_bounds__(256) void convT_bf16_kernel(const float* __restrict__ src,
                                                         unsigned short* __restrict__ dst,
                                                         int Kd, int N) {
  int i = blockIdx.x * 256 + threadIdx.x;
  const int total = Kd * N;
  const int stride = gridDim.x * 256;
  for (; i < total; i += stride) {
    const int k = i / N;
    const int n = i - k * N;
    dst[(long)n * Kd + k] = f2bf(src[i]);
  }
}

// ---------------- WMMA bf16 GEMM:  C = act( alpha * A@B + bias ) ----------------
// A: [M, Kd] bf16 row-major (lda), per-z offset sAh
// BT==1: Bm is [N, Kd] row-major (B^T / N-major) -> contiguous LDS staging
// BT==0: Bm is [Kd, N] row-major                 -> transpose-on-store staging
// Block = 256 threads (8 waves, 4x2 wave grid); tile 128 x TN, K-step 32,
// double-buffered LDS, last K-step peeled (branch-free steady-state loop).
// Each wave owns a 32 x (TN/2) C tile -> 2*(TN/32) v_wmma_f32_16x16x32_bf16 per step.
#define LDSA 40  /* padded row pitch in halfs: 40*2=80B, multiple of 16B */
#define LDSB 40

template <int BT, int TN>
__global__ __launch_bounds__(256) void gemm_bf16_kernel(
    const unsigned short* __restrict__ A, int lda, long sAh,
    const unsigned short* __restrict__ Bm, int ldb, long sBh,
    const float* __restrict__ bias,
    float* __restrict__ outF, unsigned short* __restrict__ outH,
    int ldc, long sCh, int Kd, float alpha, int act)
{
  constexpr int NB = TN / 32;   // B column-fragments per wave (2 or 4)
  __shared__ unsigned short As[2][128 * LDSA];
  __shared__ unsigned short Bs[2][TN * LDSB];

  const int hh = blockIdx.z;
  const unsigned short* Ab = A + (long)hh * sAh;
  const unsigned short* Bb = Bm + (long)hh * sBh;
  const long coff = (long)hh * sCh;

  const int m0 = blockIdx.y * 128;
  const int n0 = blockIdx.x * TN;

  const int tid  = threadIdx.x;
  const int lane = tid & 31;
  const int wave = tid >> 5;
  const int wm = (wave >> 1) * 32;        // 0,32,64,96
  const int wn = (wave & 1) * (TN / 2);   // 0, TN/2
  const int half = lane >> 4;
  const int l16  = lane & 15;

  v8f c0[NB], c1[NB];
  {
    v8f z = {};
    #pragma unroll
    for (int j = 0; j < NB; ++j) { c0[j] = z; c1[j] = z; }
  }

  // A staging: 128x32 halfs; each thread stores two b128 chunks (rows ar, ar+64)
  const int ar = tid >> 2;           // 0..63
  const int ac = (tid & 3) * 8;      // 0,8,16,24
  const unsigned short* agp0 = Ab + (long)(m0 + ar) * lda + ac;
  const unsigned short* agp1 = agp0 + (long)64 * lda;
  const int asto = ar * LDSA + ac;

  // B staging
  const unsigned short* bgp0;
  const unsigned short* bgp1 = nullptr;
  int bsto = 0, bk = 0, bn = 0;
  if (BT) {
    bgp0 = Bb + (long)(n0 + ar) * ldb + ac;   // row n0+ar, cols k
    if (TN == 128) bgp1 = bgp0 + (long)64 * ldb;
    bsto = ar * LDSB + ac;
  } else {
    bk = tid >> 3;                            // 0..31 (k)
    bn = (tid & 7) * 8;                       // 0..56 (n)
    bgp0 = Bb + (long)bk * ldb + n0 + bn;
  }

  // ---- preload tile k0=0 into buffer 0 ----
  {
    v4u a0 = *(const v4u*)agp0;
    v4u a1 = *(const v4u*)agp1;
    *(v4u*)&As[0][asto] = a0;
    *(v4u*)&As[0][64 * LDSA + asto] = a1;
    if (BT) {
      v4u b0 = *(const v4u*)bgp0;
      *(v4u*)&Bs[0][bsto] = b0;
      if (TN == 128) {
        v4u b1 = *(const v4u*)bgp1;
        *(v4u*)&Bs[0][64 * LDSB + bsto] = b1;
      }
    } else {
      union { v4u q; unsigned short s[8]; } t;
      t.q = *(const v4u*)bgp0;
      #pragma unroll
      for (int i = 0; i < 8; ++i) Bs[0][(bn + i) * LDSB + bk] = t.s[i];
    }
  }
  __syncthreads();

  // ---- fragment load + 2*NB WMMAs from one LDS buffer ----
  auto compute = [&](int bufi) {
    const unsigned short* asb = &As[bufi][0];
    const unsigned short* bsb = &Bs[bufi][0];
    union { v16bf v; v4u q[2]; } fa0, fa1, fb[NB];
    {
      const unsigned short* p = &asb[(wm + l16) * LDSA + 8 * half];
      fa0.q[0] = *(const v4u*)p;        // K in [8*half, +8)
      fa0.q[1] = *(const v4u*)(p + 16); // K in [16+8*half, +8)
      p += 16 * LDSA;
      fa1.q[0] = *(const v4u*)p;
      fa1.q[1] = *(const v4u*)(p + 16);
    }
    #pragma unroll
    for (int j = 0; j < NB; ++j) {
      const unsigned short* p = &bsb[(wn + 16 * j + l16) * LDSB + 8 * half];
      fb[j].q[0] = *(const v4u*)p;      // K in [8*half, +8)
      fb[j].q[1] = *(const v4u*)(p + 16);
    }
    #pragma unroll
    for (int j = 0; j < NB; ++j) {
      c0[j] = __builtin_amdgcn_wmma_f32_16x16x32_bf16(false, fa0.v, false, fb[j].v,
                                                      (short)0, c0[j], false, false);
      c1[j] = __builtin_amdgcn_wmma_f32_16x16x32_bf16(false, fa1.v, false, fb[j].v,
                                                      (short)0, c1[j], false, false);
    }
  };

  // ---- steady-state: unconditional prefetch + compute + store (no branches) ----
  int buf = 0;
  int k0 = 0;
  for (; k0 + 32 < Kd; k0 += 32) {
    agp0 += 32; agp1 += 32;
    v4u a0n = *(const v4u*)agp0;
    v4u a1n = *(const v4u*)agp1;
    v4u b0n, b1n;
    if (BT) {
      bgp0 += 32; b0n = *(const v4u*)bgp0;
      if (TN == 128) { bgp1 += 32; b1n = *(const v4u*)bgp1; }
    } else {
      bgp0 += (long)32 * ldb; b0n = *(const v4u*)bgp0;
    }

    compute(buf);

    unsigned short* asn = &As[buf ^ 1][0];
    unsigned short* bsn = &Bs[buf ^ 1][0];
    *(v4u*)&asn[asto] = a0n;
    *(v4u*)&asn[64 * LDSA + asto] = a1n;
    if (BT) {
      *(v4u*)&bsn[bsto] = b0n;
      if (TN == 128) *(v4u*)&bsn[64 * LDSB + bsto] = b1n;
    } else {
      union { v4u q; unsigned short s[8]; } t; t.q = b0n;
      #pragma unroll
      for (int i = 0; i < 8; ++i) bsn[(bn + i) * LDSB + bk] = t.s[i];
    }
    __syncthreads();
    buf ^= 1;
  }
  // ---- peeled final K-step ----
  compute(buf);

  // ---- epilogue: VGPR r, lanes 0-15: M=r, lanes 16-31: M=r+8; N=l16 ----
  #pragma unroll
  for (int j = 0; j < NB; ++j) {
    #pragma unroll
    for (int r = 0; r < 8; ++r) {
      const int rowA = m0 + wm + r + 8 * half;
      const int rowB = rowA + 16;
      const int col  = n0 + wn + 16 * j + l16;
      float vA = c0[j][r] * alpha;
      float vB = c1[j][r] * alpha;
      if (bias) { const float bb = bias[col]; vA += bb; vB += bb; }
      if (act)  { vA = gelu_exact(vA); vB = gelu_exact(vB); }
      const long iA = coff + (long)rowA * ldc + col;
      const long iB = coff + (long)rowB * ldc + col;
      if (outF) { outF[iA] = vA; outF[iB] = vB; }
      if (outH) { outH[iA] = f2bf(vA); outH[iB] = f2bf(vB); }
    }
  }
}

// ---------------- embedding + LayerNorm ----------------
__global__ __launch_bounds__(256) void embed_ln_kernel(
    const int* __restrict__ ids,
    const float* __restrict__ wemb, const float* __restrict__ pemb,
    const float* __restrict__ temb,
    const float* __restrict__ lw, const float* __restrict__ lb,
    float* __restrict__ outF, unsigned short* __restrict__ outH)
{
  __shared__ float red[256];
  const long row = blockIdx.x;
  const int s = (int)(row % SS);
  const int id = ids[row];
  const int tid = threadIdx.x;

  float e[3]; float lsum = 0.f;
  #pragma unroll
  for (int i = 0; i < 3; ++i) {
    const int d = tid + 256 * i;
    e[i] = wemb[(long)id * DD + d] + pemb[(long)s * DD + d] + temb[d];
    lsum += e[i];
  }
  red[tid] = lsum; __syncthreads();
  for (int off = 128; off > 0; off >>= 1) { if (tid < off) red[tid] += red[tid + off]; __syncthreads(); }
  const float mean = red[0] * (1.0f / DD); __syncthreads();

  float lsq = 0.f;
  #pragma unroll
  for (int i = 0; i < 3; ++i) { const float d = e[i] - mean; lsq += d * d; }
  red[tid] = lsq; __syncthreads();
  for (int off = 128; off > 0; off >>= 1) { if (tid < off) red[tid] += red[tid + off]; __syncthreads(); }
  const float inv = rsqrtf(red[0] * (1.0f / DD) + EPSF);

  #pragma unroll
  for (int i = 0; i < 3; ++i) {
    const int d = tid + 256 * i;
    const float o = (e[i] - mean) * inv * lw[d] + lb[d];
    outF[row * DD + d] = o;
    outH[row * DD + d] = f2bf(o);
  }
}

// ---------------- residual + LayerNorm ----------------
__global__ __launch_bounds__(256) void resid_ln_kernel(
    const float* __restrict__ x, const float* __restrict__ y,
    const float* __restrict__ lw, const float* __restrict__ lb,
    float* __restrict__ outF, unsigned short* __restrict__ outH)
{
  __shared__ float red[256];
  const long row = blockIdx.x;
  const int tid = threadIdx.x;

  float e[3]; float lsum = 0.f;
  #pragma unroll
  for (int i = 0; i < 3; ++i) {
    const int d = tid + 256 * i;
    e[i] = x[row * DD + d] + y[row * DD + d];
    lsum += e[i];
  }
  red[tid] = lsum; __syncthreads();
  for (int off = 128; off > 0; off >>= 1) { if (tid < off) red[tid] += red[tid + off]; __syncthreads(); }
  const float mean = red[0] * (1.0f / DD); __syncthreads();

  float lsq = 0.f;
  #pragma unroll
  for (int i = 0; i < 3; ++i) { const float d = e[i] - mean; lsq += d * d; }
  red[tid] = lsq; __syncthreads();
  for (int off = 128; off > 0; off >>= 1) { if (tid < off) red[tid] += red[tid + off]; __syncthreads(); }
  const float inv = rsqrtf(red[0] * (1.0f / DD) + EPSF);

  #pragma unroll
  for (int i = 0; i < 3; ++i) {
    const int d = tid + 256 * i;
    const float o = (e[i] - mean) * inv * lw[d] + lb[d];
    if (outF) outF[row * DD + d] = o;
    if (outH) outH[row * DD + d] = f2bf(o);
  }
}

// ---------------- masked softmax (exact reference semantics) ----------------
__global__ __launch_bounds__(128) void softmax_kernel(
    const float* __restrict__ scores, const int* __restrict__ mask,
    unsigned short* __restrict__ P)
{
  __shared__ float red[128];
  const int q = blockIdx.x;
  const int h = blockIdx.y;
  const float* a = scores + ((long)h * SS + q) * SS;
  unsigned short* p = P + ((long)h * SS + q) * SS;
  const int tid = threadIdx.x;

  float mv[4], mk[4];
  float mx = -3.4e38f;
  #pragma unroll
  for (int i = 0; i < 4; ++i) {
    const int c = tid + 128 * i;
    mk[i] = (float)mask[c];
    mv[i] = a[c] * mk[i];
    mx = fmaxf(mx, mv[i]);
  }
  red[tid] = mx; __syncthreads();
  for (int off = 64; off > 0; off >>= 1) { if (tid < off) red[tid] = fmaxf(red[tid], red[tid + off]); __syncthreads(); }
  mx = red[0]; __syncthreads();

  float ex[4]; float sum = 0.f;
  #pragma unroll
  for (int i = 0; i < 4; ++i) { ex[i] = expf(mv[i] - mx) * mk[i]; sum += ex[i]; }
  red[tid] = sum; __syncthreads();
  for (int off = 64; off > 0; off >>= 1) { if (tid < off) red[tid] += red[tid + off]; __syncthreads(); }
  sum = red[0];
  sum += (sum == 0.f) ? 1.f : 0.f;   // zero-sum guard
  const float inv = 1.f / sum;

  #pragma unroll
  for (int i = 0; i < 4; ++i) {
    const int c = tid + 128 * i;
    p[c] = f2bf(ex[i] * inv);
  }
}

// ---------------- host-side helpers ----------------
static void launch_convT(hipStream_t st, const float* src, unsigned short* dst, int Kd, int N) {
  int blocks = (Kd * N + 255) / 256;
  if (blocks > 8192) blocks = 8192;
  convT_bf16_kernel<<<blocks, 256, 0, st>>>(src, dst, Kd, N);
}

static void launch_gemm(hipStream_t st,
                        const unsigned short* A, int lda, long sAh,
                        const unsigned short* B, int ldb, long sBh, int btrans,
                        const float* bias, float* outF, unsigned short* outH,
                        int ldc, long sCh, int M, int N, int Kd,
                        float alpha, int act, int zdim) {
  if (btrans) {
    dim3 grid(N / 128, M / 128, zdim);
    gemm_bf16_kernel<1, 128><<<grid, 256, 0, st>>>(A, lda, sAh, B, ldb, sBh,
                                                   bias, outF, outH, ldc, sCh, Kd, alpha, act);
  } else {
    dim3 grid(N / 64, M / 128, zdim);
    gemm_bf16_kernel<0, 64><<<grid, 256, 0, st>>>(A, lda, sAh, B, ldb, sBh,
                                                  bias, outF, outH, ldc, sCh, Kd, alpha, act);
  }
}

extern "C" void kernel_launch(void* const* d_in, const int* in_sizes, int n_in,
                              void* d_out, int out_size, void* d_ws, size_t ws_size,
                              hipStream_t stream) {
  (void)in_sizes; (void)n_in; (void)out_size; (void)ws_size;

  const int*   ids   = (const int*)d_in[0];
  const int*   amask = (const int*)d_in[1];
  const float* wemb  = (const float*)d_in[2];
  const float* pemb  = (const float*)d_in[3];
  const float* temb  = (const float*)d_in[4];
  const float* elnw  = (const float*)d_in[5];
  const float* elnb  = (const float*)d_in[6];
  const float* Wq    = (const float*)d_in[7];
  const float* bq    = (const float*)d_in[8];
  const float* Wk    = (const float*)d_in[9];
  const float* bk    = (const float*)d_in[10];
  const float* Wv    = (const float*)d_in[11];
  const float* bv    = (const float*)d_in[12];
  const float* Wo    = (const float*)d_in[13];
  const float* bo    = (const float*)d_in[14];
  const float* l1w   = (const float*)d_in[15];
  const float* l1b   = (const float*)d_in[16];
  const float* Wf1   = (const float*)d_in[17];
  const float* bf1   = (const float*)d_in[18];
  const float* Wf2   = (const float*)d_in[19];
  const float* bf2   = (const float*)d_in[20];
  const float* l2w   = (const float*)d_in[21];
  const float* l2b   = (const float*)d_in[22];

  // ---- carve workspace ----
  char* ws = (char*)d_ws;
  size_t off = 0;
  auto carve = [&](size_t bytes) -> void* {
    void* p = ws + off;
    off += (bytes + 255) & ~(size_t)255;
    return p;
  };
  float*          hF    = (float*)carve(sizeof(float) * (size_t)MM * DD);
  float*          h1F   = (float*)carve(sizeof(float) * (size_t)MM * DD);
  float*          tmpF  = (float*)carve(sizeof(float) * (size_t)MM * DD);
  unsigned short* actH  = (unsigned short*)carve(2ull * MM * DD);
  unsigned short* qH    = (unsigned short*)carve(2ull * MM * HK);
  unsigned short* kH    = (unsigned short*)carve(2ull * MM * HK);
  unsigned short* vH    = (unsigned short*)carve(2ull * MM * HK);
  unsigned short* ctxH  = (unsigned short*)carve(2ull * MM * HK);
  unsigned short* ffnH  = (unsigned short*)carve(2ull * MM * FF);
  float*          scF   = (float*)carve(sizeof(float) * (size_t)HH * SS * SS);
  unsigned short* pH    = (unsigned short*)carve(2ull * HH * SS * SS);
  unsigned short* wbufT = (unsigned short*)carve(2ull * DD * FF);

  // ---- embeddings + LN ----
  embed_ln_kernel<<<MM, 256, 0, stream>>>(ids, wemb, pemb, temb, elnw, elnb, hF, actH);

  // ---- encoder layers ----
  for (int l = 0; l < LL; ++l) {
    const float* wq_l  = Wq  + (size_t)l * DD * HK;
    const float* wk_l  = Wk  + (size_t)l * DD * HK;
    const float* wv_l  = Wv  + (size_t)l * DD * HK;
    const float* wo_l  = Wo  + (size_t)l * HK * DD;
    const float* wf1_l = Wf1 + (size_t)l * DD * FF;
    const float* wf2_l = Wf2 + (size_t)l * FF * DD;
    const float* bq_l  = bq  + (size_t)l * HK;
    const float* bk_l  = bk  + (size_t)l * HK;
    const float* bv_l  = bv  + (size_t)l * HK;
    const float* bo_l  = bo  + (size_t)l * DD;
    const float* bf1_l = bf1 + (size_t)l * FF;
    const float* bf2_l = bf2 + (size_t)l * DD;
    const float* l1w_l = l1w + (size_t)l * DD;
    const float* l1b_l = l1b + (size_t)l * DD;
    const float* l2w_l = l2w + (size_t)l * DD;
    const float* l2b_l = l2b + (size_t)l * DD;

    // Q/K/V projections: [8192,768] @ [768,768] + bias -> bf16 (weights pre-transposed)
    launch_convT(stream, wq_l, wbufT, DD, HK);
    launch_gemm(stream, actH, DD, 0, wbufT, DD, 0, 1, bq_l, nullptr, qH,
                HK, 0, MM, HK, DD, 1.0f, 0, 1);
    launch_convT(stream, wk_l, wbufT, DD, HK);
    launch_gemm(stream, actH, DD, 0, wbufT, DD, 0, 1, bk_l, nullptr, kH,
                HK, 0, MM, HK, DD, 1.0f, 0, 1);
    launch_convT(stream, wv_l, wbufT, DD, HK);
    launch_gemm(stream, actH, DD, 0, wbufT, DD, 0, 1, bv_l, nullptr, vH,
                HK, 0, MM, HK, DD, 1.0f, 0, 1);

    // attention, looped over batch element b (keeps S*S scratch to one batch)
    for (int b = 0; b < BB; ++b) {
      const unsigned short* qb = qH + (size_t)b * SS * HK;
      const unsigned short* kb = kH + (size_t)b * SS * HK;
      const unsigned short* vb = vH + (size_t)b * SS * HK;
      unsigned short*       cb = ctxH + (size_t)b * SS * HK;

      // scores = (1/8) * Q @ K^T per head (z), [512,512], Kd=64; K-matrix already N-major
      launch_gemm(stream, qb, HK, (long)KK, kb, HK, (long)KK, 1, nullptr,
                  scF, nullptr, SS, (long)SS * SS, SS, SS, KK, 0.125f, 0, HH);

      // masked softmax -> P (bf16)
      softmax_kernel<<<dim3(SS, HH), 128, 0, stream>>>(scF, amask + (size_t)b * SS, pH);

      // context = P @ V per head, [512,64], Kd=512 (V in [Kd,N] layout -> BT=0 path)
      launch_gemm(stream, pH, SS, (long)SS * SS, vb, HK, (long)VV, 0, nullptr,
                  nullptr, cb, HK, (long)VV, SS, VV, SS, 1.0f, 0, HH);
    }

    // output projection: ctx @ Wo + bo -> f32 tmp
    launch_convT(stream, wo_l, wbufT, HK, DD);
    launch_gemm(stream, ctxH, HK, 0, wbufT, HK, 0, 1, bo_l, tmpF, nullptr,
                DD, 0, MM, DD, HK, 1.0f, 0, 1);

    // h1 = LN(h + attn_out)
    resid_ln_kernel<<<MM, 256, 0, stream>>>(hF, tmpF, l1w_l, l1b_l, h1F, actH);

    // FFN1: h1 @ Wf1 + bf1, GELU -> bf16
    launch_convT(stream, wf1_l, wbufT, DD, FF);
    launch_gemm(stream, actH, DD, 0, wbufT, DD, 0, 1, bf1_l, nullptr, ffnH,
                FF, 0, MM, FF, DD, 1.0f, 1, 1);

    // FFN2: ffn @ Wf2 + bf2 -> f32 tmp
    launch_convT(stream, wf2_l, wbufT, FF, DD);
    launch_gemm(stream, ffnH, FF, 0, wbufT, FF, 0, 1, bf2_l, tmpF, nullptr,
                DD, 0, MM, DD, FF, 1.0f, 0, 1);

    // h = LN(h1 + ffn_out); last layer writes straight to d_out
    float* outTarget = (l == LL - 1) ? (float*)d_out : hF;
    resid_ln_kernel<<<MM, 256, 0, stream>>>(h1F, tmpF, l2w_l, l2b_l, outTarget, actH);
  }
}
